// NNUE_32152125178615
// MI455X (gfx1250) — matrix operand order repaired and checked by reading, not compiled, not run on hardware
//
#include <hip/hip_runtime.h>

typedef _Float16 half_t;
typedef __attribute__((ext_vector_type(16))) _Float16 v16h;
typedef __attribute__((ext_vector_type(4)))  _Float16 h4;
typedef __attribute__((ext_vector_type(8)))  float    v8f;

#define FDIM   768
#define L1DIM  1024
#define BATCH  65536
#define ROWS   64
#define KSTEPS 24          // 768 / 32

// dynamic LDS layout (bytes) -- double-buffered panels
#define OFF_B0   0          // 64 KB : B panel buffer 0
#define OFF_B1   65536      // 64 KB : B panel buffer 1
#define OFF_A0   131072     // 4 KB  : A panel buffer 0
#define OFF_A1   135168     // 4 KB  : A panel buffer 1
#define OFF_BIN  139264     // 4 KB  : b_in f32
#define OFF_WH   143360     // 32 KB : W_h f32 [8][1024]
#define OFF_OUTP 176128     // 2 KB  : per-row 8-head partials
#define OFF_RSUM 178176     // 256 B
#define OFF_RPSQ 178432     // 256 B
#define SMEM_BYTES 178688

// async DMA: global -> LDS, 16B per lane, tracked by ASYNCcnt
__device__ __forceinline__ void async_copy_b128(unsigned lds_addr, const void* gaddr) {
    asm volatile("global_load_async_to_lds_b128 %0, %1, off"
                 :: "v"(lds_addr), "v"((unsigned long long)(uintptr_t)gaddr)
                 : "memory");
}
#define WAIT_ASYNC(n) asm volatile("s_wait_asynccnt %0" :: "i"(n) : "memory")

// ---------------------------------------------------------------------------
// Pack W_in (f32 [L1,F]; h = x @ W_in.T  =>  B[k=f][n=l] = W_in[l][f]) into
// f16 WMMA B-fragment order: frag fi = kstep*64 + coltile, lane-major
// (32 B per lane).  B 32x16 f16 order used here (contiguous k): N = lane%16, K = elem + 16*(lane/16).
// ---------------------------------------------------------------------------
__global__ void nnue_pack_w(const float* __restrict__ W_in,
                            half_t* __restrict__ packed) {
    int idx    = blockIdx.x * 256 + threadIdx.x;   // 0 .. 786431
    int within = idx & 511;
    int fi     = idx >> 9;
    int L      = within >> 4;                      // lane
    int e      = within & 15;                      // element within v16h
    int k      = fi >> 6;                          // K-step
    int c      = fi & 63;                          // col tile
    int n      = c * 16 + (L & 15);                // hidden index l
    int kk     = k * 32 + e + ((L >> 4) << 4);     // feature index f
    packed[idx] = (half_t)W_in[n * FDIM + kk];
}

// ---------------------------------------------------------------------------
// Fused NNUE: 64 rows/WG, 1024 hidden cols in registers, async double-buffered
// weight panels, clipped ReLU + 8-head GEMV + bucket + psqt in one pass.
// ---------------------------------------------------------------------------
__global__ __launch_bounds__(512, 1) void nnue_fused(
    const float*  __restrict__ x,
    const half_t* __restrict__ Wp,      // packed W_in f16 (workspace)
    const float*  __restrict__ b_in,
    const float*  __restrict__ W_h,
    const float*  __restrict__ b_h,
    const float*  __restrict__ W_psqt,
    float*        __restrict__ out) {
    extern __shared__ char smem[];
    float* binl = (float*)(smem + OFF_BIN);
    float* whl  = (float*)(smem + OFF_WH);
    float* outp = (float*)(smem + OFF_OUTP);
    float* rsum = (float*)(smem + OFF_RSUM);
    float* rpsq = (float*)(smem + OFF_RPSQ);

    const int tid      = threadIdx.x;
    const int row_base = blockIdx.x * ROWS;

    // preload small tensors, init reductions
    for (int i = tid; i < L1DIM; i += 512)     binl[i] = b_in[i];
    for (int i = tid; i < 8 * L1DIM; i += 512) whl[i]  = W_h[i];
    outp[tid] = b_h[tid & 7];
    if (tid < ROWS) { rsum[tid] = 0.0f; rpsq[tid] = 0.0f; }
    __syncthreads();

    // --- A-staging geometry: thread -> (row, 4 consecutive features) ---
    // A 16x32 f16 frag order used here (contiguous k, matches B): M = lane%16, K = e + 16*(lane/16).
    const int arow = tid >> 3;                    // 0..63
    const int g    = tid & 7;                     // feature group of 4
    const int b3   = (g >> 1) & 1;
    const int b4   = (g >> 2) & 1;
    const int aL   = (arow & 15) + (b4 << 4);     // target lane
    const int ae0  = ((g & 1) << 2) + (b3 << 3);  // first elem
    const int art  = arow >> 4;                   // row tile
    const int adst = art * 1024 + aL * 32 + ae0 * 2;   // byte off in A panel
    const float4* xrow = (const float4*)(x + (size_t)(row_base + arow) * FDIM);
    const float4* wpq  = (const float4*)W_psqt;

    // --- compute geometry: wave owns 4 row-tiles x 4 col-tiles ---
    const int wave = tid >> 5;            // 0..15
    const int lane = tid & 31;

    v8f acc[4][4];
    const v8f vzero = {0.f, 0.f, 0.f, 0.f, 0.f, 0.f, 0.f, 0.f};
#pragma unroll
    for (int i = 0; i < 4; ++i)
#pragma unroll
        for (int j = 0; j < 4; ++j) acc[i][j] = vzero;

    float sx = 0.0f, sp = 0.0f;

    // stage step s into buffer s&1: async 64KB B panel + f32->f16 A panel
    auto stage = [&](int s) {
        float4 v  = xrow[s * 8 + g];
        float4 wv = wpq[s * 8 + g];
        sx += v.x + v.y + v.z + v.w;
        sp += v.x * wv.x + v.y * wv.y + v.z * wv.z + v.w * wv.w;
        h4 hv;
        hv.x = (half_t)v.x; hv.y = (half_t)v.y;
        hv.z = (half_t)v.z; hv.w = (half_t)v.w;
        char* ab = smem + ((s & 1) ? OFF_A1 : OFF_A0);
        *(h4*)(ab + adst) = hv;

        const char* gs = (const char*)Wp + (size_t)s * 65536 + tid * 16;
        unsigned    ls = (unsigned)(uintptr_t)(smem + ((s & 1) ? OFF_B1 : OFF_B0) + tid * 16);
#pragma unroll
        for (int i = 0; i < 8; ++i)
            async_copy_b128(ls + i * 8192, gs + i * 8192);
        if (s + 1 < KSTEPS) __builtin_prefetch(&xrow[(s + 1) * 8 + g], 0, 1);
    };

    stage(0);
#pragma unroll 1
    for (int ks = 0; ks < KSTEPS; ++ks) {
        if (ks + 1 < KSTEPS) { stage(ks + 1); WAIT_ASYNC(8); }
        else                 { WAIT_ASYNC(0); }
        __syncthreads();                       // panel ks visible WG-wide

        const char* ab = smem + ((ks & 1) ? OFF_A1 : OFF_A0);
        const char* bb = smem + ((ks & 1) ? OFF_B1 : OFF_B0);
        v16h a0 = *(const v16h*)(ab +    0 + lane * 32);
        v16h a1 = *(const v16h*)(ab + 1024 + lane * 32);
        v16h a2 = *(const v16h*)(ab + 2048 + lane * 32);
        v16h a3 = *(const v16h*)(ab + 3072 + lane * 32);
#pragma unroll
        for (int ct = 0; ct < 4; ++ct) {
            v16h b = *(const v16h*)(bb + (wave * 4 + ct) * 1024 + lane * 32);
            acc[0][ct] = __builtin_amdgcn_wmma_f32_16x16x32_f16(false, a0, false, b, (short)0, acc[0][ct], false, false);
            acc[1][ct] = __builtin_amdgcn_wmma_f32_16x16x32_f16(false, a1, false, b, (short)0, acc[1][ct], false, false);
            acc[2][ct] = __builtin_amdgcn_wmma_f32_16x16x32_f16(false, a2, false, b, (short)0, acc[2][ct], false, false);
            acc[3][ct] = __builtin_amdgcn_wmma_f32_16x16x32_f16(false, a3, false, b, (short)0, acc[3][ct], false, false);
        }
        __syncthreads();                       // done reading before reuse
    }

    // bucket / psqt reductions (8 staging threads per row)
    atomicAdd(&rsum[arow], sx);
    atomicAdd(&rpsq[arow], sp);

    // --- epilogue: clipped ReLU + 8-head GEMV ---
    // C layout: M = r + 8*(lane/16), N = lane%16
    const int lmod  = lane & 15;
    const int rowhi = (lane >> 4) << 3;       // 0 or 8
    float bb2[4];
    float wj2[4][8];
#pragma unroll
    for (int ct = 0; ct < 4; ++ct) {
        int l = (wave * 4 + ct) * 16 + lmod;
        bb2[ct] = binl[l];
#pragma unroll
        for (int j = 0; j < 8; ++j) wj2[ct][j] = whl[j * L1DIM + l];
    }
#pragma unroll
    for (int rt = 0; rt < 4; ++rt) {
        float part[8][8];
#pragma unroll
        for (int r = 0; r < 8; ++r)
#pragma unroll
            for (int j = 0; j < 8; ++j) part[r][j] = 0.0f;
#pragma unroll
        for (int ct = 0; ct < 4; ++ct)
#pragma unroll
            for (int r = 0; r < 8; ++r) {
                float h = fminf(fmaxf(acc[rt][ct][r] + bb2[ct], 0.0f), 1.0f);
#pragma unroll
                for (int j = 0; j < 8; ++j) part[r][j] += h * wj2[ct][j];
            }
#pragma unroll
        for (int r = 0; r < 8; ++r) {
            int rw = rt * 16 + rowhi + r;
#pragma unroll
            for (int j = 0; j < 8; ++j)
                atomicAdd(&outp[rw * 8 + j], part[r][j]);
        }
    }
    __syncthreads();

    // final: pick bucket head, add psqt
    if (tid < ROWS) {
        float s      = rsum[tid];
        int   bucket = (int)((s - 1.0f) * 0.25f);
        bucket = bucket < 0 ? 0 : (bucket > 7 ? 7 : bucket);
        out[row_base + tid] = outp[tid * 8 + bucket] + rpsq[tid];
    }
}

extern "C" void kernel_launch(void* const* d_in, const int* in_sizes, int n_in,
                              void* d_out, int out_size, void* d_ws, size_t ws_size,
                              hipStream_t stream) {
    (void)in_sizes; (void)n_in; (void)out_size; (void)ws_size;
    const float* x      = (const float*)d_in[0];
    const float* W_in   = (const float*)d_in[1];
    const float* b_in   = (const float*)d_in[2];
    const float* W_h    = (const float*)d_in[3];
    const float* b_h    = (const float*)d_in[4];
    const float* W_psqt = (const float*)d_in[5];
    float*  out = (float*)d_out;
    half_t* Wp  = (half_t*)d_ws;   // 1.5 MB packed f16 W_in

    hipFuncSetAttribute(reinterpret_cast<const void*>(&nnue_fused),
                        hipFuncAttributeMaxDynamicSharedMemorySize, SMEM_BYTES);

    nnue_pack_w<<<(L1DIM * FDIM) / 256, 256, 0, stream>>>(W_in, Wp);
    nnue_fused<<<BATCH / ROWS, 512, SMEM_BYTES, stream>>>(
        x, Wp, b_in, W_h, b_h, W_psqt, out);
}
